// Attention_17231408791568
// MI455X (gfx1250) — matrix operand order repaired and checked by reading, not compiled, not run on hardware
//
#include <hip/hip_runtime.h>
#include <cstdint>

// ---------------------------------------------------------------------------
// MI455X / gfx1250 dual-branch attention, bf16 WMMA everywhere.
// Pipeline:
//   0) convert se/de f32->bf16; convert+transpose all weights f32->bf16 [N,K]
//   1) GEMM (bf16 WMMA, double-buffered async global->LDS DMA overlapping
//      the WMMA stream): qkv, v_de projections
//   2) repack: q (*SCALE), k as [B,H,N,32]; v_se, v_de transposed [B,H,32,N]
//   3) flash attention (online softmax), 2 value branches sharing P
//   4) GEMM + bias (f32 out) -> d_out (out_se then out_de, concatenated)
// ---------------------------------------------------------------------------

typedef __bf16 bf16;
typedef __attribute__((ext_vector_type(16))) bf16  v16bf;
typedef __attribute__((ext_vector_type(8)))  bf16  v8bf;
typedef __attribute__((ext_vector_type(8)))  float v8f;

#define B_ 2
#define N_ 2048
#define C_ 512
#define H_ 16
#define D_ 32
#define SCALE_ 0.17677669529663689f   // 32^-0.5

union Frag { v16bf v; v8bf h[2]; };

static __device__ __forceinline__ v8f wmma_bf16(v16bf a, v16bf b, v8f c) {
  // 8 args: (neg_a, A, neg_b, B, c_mod, C, reuse_a, reuse_b)
  return __builtin_amdgcn_wmma_f32_16x16x32_bf16(false, a, false, b, (short)0, c,
                                                 false, false);
}

// Async DMA: global memory -> LDS, 16 bytes per lane, tracked by ASYNCcnt.
static __device__ __forceinline__ void async_g2l_b128(uint32_t lds_addr,
                                                      const bf16* gptr) {
  uint64_t ga = (uint64_t)(uintptr_t)gptr;
  asm volatile("global_load_async_to_lds_b128 %0, %1, off"
               :: "v"(lds_addr), "v"(ga) : "memory");
}
static __device__ __forceinline__ void wait_async0() {
  asm volatile("s_wait_asynccnt 0x0" ::: "memory");
}

// ------------------------- conversion kernels ------------------------------

__global__ void k_cvt(const float* __restrict__ in, bf16* __restrict__ out, int n) {
  int i = blockIdx.x * 256 + threadIdx.x;
  if (i < n) out[i] = (bf16)in[i];
}

// W [K,Nn] row-major f32  ->  Wt [Nn,K] row-major bf16
__global__ void k_cvt_t(const float* __restrict__ W, bf16* __restrict__ Wt,
                        int K, int Nn) {
  int i = blockIdx.x * 256 + threadIdx.x;
  if (i < K * Nn) {
    int kk = i / Nn, n = i - kk * Nn;
    Wt[(size_t)n * K + kk] = (bf16)W[i];
  }
}

// ------------------------- tiled WMMA GEMM ---------------------------------
// C[M,Nn] = A[M,K] @ Bt[Nn,K]^T (+bias). 128x128 block, BK=32, 8 waves,
// each wave 64x32 (4x2 accumulator tiles). Double-buffered LDS filled by
// async DMA; tile i+1 DMA overlaps tile i WMMA.

__global__ __launch_bounds__(256) void k_gemm(
    const bf16* __restrict__ A, const bf16* __restrict__ Bt,
    int M, int Nn, int K,
    const float* __restrict__ bias,
    float* __restrict__ outF, bf16* __restrict__ outB)
{
  __shared__ __align__(16) bf16 As[2][128][32];   // 8KB x2
  __shared__ __align__(16) bf16 Bs[2][128][32];   // 8KB x2

  const int t = threadIdx.x;
  const int lane = t & 31, wav = t >> 5, half = lane >> 4, ln = lane & 15;
  const int waveM = wav >> 2, waveN = wav & 3;          // 2 x 4 wave grid
  const int bM0 = blockIdx.y * 128, bN0 = blockIdx.x * 128;
  const int lrow = t >> 1, lseg = (t & 1) * 16;          // global->LDS mapping

  v8f acc[4][2];
#pragma unroll
  for (int i = 0; i < 4; i++)
#pragma unroll
    for (int j = 0; j < 2; j++) acc[i][j] = (v8f)0.0f;

  const bf16* ga = A  + (size_t)(bM0 + lrow) * K + lseg;
  const bf16* gb = Bt + (size_t)(bN0 + lrow) * K + lseg;
  const uint32_t lA0 = (uint32_t)(uintptr_t)&As[0][lrow][lseg];
  const uint32_t lB0 = (uint32_t)(uintptr_t)&Bs[0][lrow][lseg];
  const uint32_t bufStride =
      (uint32_t)((uintptr_t)&As[1][0][0] - (uintptr_t)&As[0][0][0]);

  // prologue: DMA tile 0 into buffer 0
  async_g2l_b128(lA0,      ga);
  async_g2l_b128(lA0 + 16, ga + 8);
  async_g2l_b128(lB0,      gb);
  async_g2l_b128(lB0 + 16, gb + 8);

  int buf = 0;
  for (int k0 = 0; k0 < K; k0 += 32) {
    wait_async0();                          // this wave's tile-i DMA landed
    __syncthreads();                        // all waves' DMA landed; all waves
                                            // finished reading the other buf
    if (k0 + 32 < K) {                      // kick tile i+1 into other buffer
      uint32_t off = (buf ^ 1) ? bufStride : 0u;
      async_g2l_b128(lA0 + off,      ga + k0 + 32);
      async_g2l_b128(lA0 + off + 16, ga + k0 + 40);
      async_g2l_b128(lB0 + off,      gb + k0 + 32);
      async_g2l_b128(lB0 + off + 16, gb + k0 + 40);
      if (k0 + 64 < K) {                    // warm L2 one tile further ahead
        __builtin_prefetch(ga + k0 + 64, 0, 1);
        __builtin_prefetch(gb + k0 + 64, 0, 1);
      }
    }

    Frag aF[4], bF[2];
#pragma unroll
    for (int mi = 0; mi < 4; mi++) {        // A-frag: two 16B row segments
      int r = waveM * 64 + mi * 16 + ln;
      aF[mi].h[0] = *(const v8bf*)&As[buf][r][half * 8];
      aF[mi].h[1] = *(const v8bf*)&As[buf][r][16 + half * 8];
    }
#pragma unroll
    for (int nj = 0; nj < 2; nj++) {        // B-frag: two 16B K segments of column n
      int n = waveN * 32 + nj * 16 + ln;
      bF[nj].h[0] = *(const v8bf*)&Bs[buf][n][half * 8];
      bF[nj].h[1] = *(const v8bf*)&Bs[buf][n][16 + half * 8];
    }
#pragma unroll
    for (int mi = 0; mi < 4; mi++)
#pragma unroll
      for (int nj = 0; nj < 2; nj++)
        acc[mi][nj] = wmma_bf16(aF[mi].v, bF[nj].v, acc[mi][nj]);
    buf ^= 1;
  }

#pragma unroll
  for (int mi = 0; mi < 4; mi++)
#pragma unroll
    for (int nj = 0; nj < 2; nj++) {
      int col = bN0 + waveN * 32 + nj * 16 + ln;
      float bb = bias ? bias[col] : 0.0f;
#pragma unroll
      for (int r = 0; r < 8; r++) {
        int row = bM0 + waveM * 64 + mi * 16 + r + 8 * half;  // C-layout rows
        float v = acc[mi][nj][r] + bb;
        if (outF) outF[(size_t)row * Nn + col] = v;
        else      outB[(size_t)row * Nn + col] = (bf16)v;
      }
    }
}

// ------------------------- repack for attention ----------------------------

__global__ void k_repack(const bf16* __restrict__ qkv, const bf16* __restrict__ vde,
                         bf16* __restrict__ q, bf16* __restrict__ k,
                         bf16* __restrict__ vseT, bf16* __restrict__ vdeT)
{
  int idx = blockIdx.x * 256 + threadIdx.x;          // over B*H*N*D = 2,097,152
  if (idx >= B_ * H_ * N_ * D_) return;
  int d = idx & 31, n = (idx >> 5) & 2047, h = (idx >> 16) & 15, b = idx >> 20;
  size_t src = (size_t)(b * N_ + n) * (3 * C_) + h * 32 + d;
  q[idx] = (bf16)((float)qkv[src] * SCALE_);         // fold softmax scale into Q
  k[idx] = qkv[src + C_];
  size_t tidx = ((size_t)((b * H_ + h) * D_ + d)) * N_ + n;  // [B,H,D,N]
  vseT[tidx] = qkv[src + 2 * C_];
  vdeT[tidx] = vde[(size_t)(b * N_ + n) * C_ + h * 32 + d];
}

// ------------------------- flash attention ---------------------------------
// grid (N/128, B*H), 256 threads = 8 waves; each wave owns 16 query rows.
// Online softmax; P bounced via per-wave LDS tile (C-layout -> A-layout).

__global__ __launch_bounds__(256) void k_attn(
    const bf16* __restrict__ q, const bf16* __restrict__ k,
    const bf16* __restrict__ vseT, const bf16* __restrict__ vdeT,
    bf16* __restrict__ attse, bf16* __restrict__ attde)
{
  __shared__ __align__(16) bf16 Pl[8][16][32];       // per-wave P tile (1KB)

  const int t = threadIdx.x, lane = t & 31, wav = t >> 5;
  const int half = lane >> 4, ln = lane & 15;
  const int bh = blockIdx.y, b = bh >> 4, h = bh & 15;
  const int qrow = blockIdx.x * 128 + wav * 16;

  const bf16* qh = q    + (size_t)bh * N_ * D_;
  const bf16* kh = k    + (size_t)bh * N_ * D_;
  const bf16* vs = vseT + (size_t)bh * D_ * N_;
  const bf16* vd = vdeT + (size_t)bh * D_ * N_;

  Frag aQ;                                           // Q A-frag, resident
  { const bf16* p = qh + (size_t)(qrow + ln) * D_;
    aQ.h[0] = *(const v8bf*)(p + half * 8);
    aQ.h[1] = *(const v8bf*)(p + 16 + half * 8); }

  float mrun[8], lrun[8];
#pragma unroll
  for (int r = 0; r < 8; r++) { mrun[r] = -3.0e38f; lrun[r] = 0.0f; }
  v8f ose0 = (v8f)0.0f, ose1 = (v8f)0.0f, ode0 = (v8f)0.0f, ode1 = (v8f)0.0f;

  for (int kb = 0; kb < N_; kb += 32) {
    Frag bK0, bK1;                                   // K^T B-frags (table order)
    { const bf16* p = kh + (size_t)(kb + ln) * D_ + half * 8;
      bK0.h[0] = *(const v8bf*)p; bK0.h[1] = *(const v8bf*)(p + 16); }
    { const bf16* p = kh + (size_t)(kb + 16 + ln) * D_ + half * 8;
      bK1.h[0] = *(const v8bf*)p; bK1.h[1] = *(const v8bf*)(p + 16); }
    v8f z = (v8f)0.0f;
    v8f s0 = wmma_bf16(aQ.v, bK0.v, z);              // S cols 0..15
    v8f s1 = wmma_bf16(aQ.v, bK1.v, z);              // S cols 16..31

    float p0[8], p1[8], corr[8];
#pragma unroll
    for (int r = 0; r < 8; r++) {                    // row = qrow + r + 8*half
      float bm = fmaxf(s0[r], s1[r]);
#pragma unroll
      for (int m = 1; m < 16; m <<= 1) bm = fmaxf(bm, __shfl_xor(bm, m, 32));
      float mn = fmaxf(mrun[r], bm);
      float c  = __expf(mrun[r] - mn);
      float e0 = __expf(s0[r] - mn);
      float e1 = __expf(s1[r] - mn);
      float rs = e0 + e1;
#pragma unroll
      for (int m = 1; m < 16; m <<= 1) rs += __shfl_xor(rs, m, 32);
      lrun[r] = lrun[r] * c + rs;
      mrun[r] = mn;
      corr[r] = c; p0[r] = e0; p1[r] = e1;
    }
#pragma unroll
    for (int r = 0; r < 8; r++) {
      ose0[r] *= corr[r]; ose1[r] *= corr[r];
      ode0[r] *= corr[r]; ode1[r] *= corr[r];
      Pl[wav][r + 8 * half][ln]      = (bf16)p0[r];  // C-layout -> row-major
      Pl[wav][r + 8 * half][16 + ln] = (bf16)p1[r];
    }
    // same-wave DS ordering; drain DScnt before re-reading the tile
    asm volatile("s_wait_dscnt 0x0" ::: "memory");

    Frag aP;                                         // P A-frag from LDS
    aP.h[0] = *(const v8bf*)&Pl[wav][ln][half * 8];
    aP.h[1] = *(const v8bf*)&Pl[wav][ln][16 + half * 8];

    Frag bV;                                         // V^T rows are contiguous
    { const bf16* p = vs + (size_t)ln * N_ + kb + half * 8;
      bV.h[0] = *(const v8bf*)p; bV.h[1] = *(const v8bf*)(p + 16); }
    ose0 = wmma_bf16(aP.v, bV.v, ose0);
    { const bf16* p = vs + (size_t)(16 + ln) * N_ + kb + half * 8;
      bV.h[0] = *(const v8bf*)p; bV.h[1] = *(const v8bf*)(p + 16); }
    ose1 = wmma_bf16(aP.v, bV.v, ose1);
    { const bf16* p = vd + (size_t)ln * N_ + kb + half * 8;
      bV.h[0] = *(const v8bf*)p; bV.h[1] = *(const v8bf*)(p + 16); }
    ode0 = wmma_bf16(aP.v, bV.v, ode0);
    { const bf16* p = vd + (size_t)(16 + ln) * N_ + kb + half * 8;
      bV.h[0] = *(const v8bf*)p; bV.h[1] = *(const v8bf*)(p + 16); }
    ode1 = wmma_bf16(aP.v, bV.v, ode1);
  }

#pragma unroll
  for (int r = 0; r < 8; r++) {
    float inv = 1.0f / lrun[r];
    int gn = qrow + r + 8 * half;
    size_t base = ((size_t)(b * N_ + gn)) * C_ + h * 32;
    attse[base + ln]      = (bf16)(ose0[r] * inv);
    attse[base + 16 + ln] = (bf16)(ose1[r] * inv);
    attde[base + ln]      = (bf16)(ode0[r] * inv);
    attde[base + 16 + ln] = (bf16)(ode1[r] * inv);
  }
}

// ------------------------- host-side orchestration -------------------------

extern "C" void kernel_launch(void* const* d_in, const int* in_sizes, int n_in,
                              void* d_out, int out_size, void* d_ws, size_t ws_size,
                              hipStream_t stream)
{
  (void)in_sizes; (void)n_in; (void)out_size; (void)ws_size;
  const float* se   = (const float*)d_in[0];
  const float* de   = (const float*)d_in[1];
  const float* Wqkv = (const float*)d_in[2];
  const float* Wv   = (const float*)d_in[3];
  const float* Wpse = (const float*)d_in[4];
  const float* bpse = (const float*)d_in[5];
  const float* Wpde = (const float*)d_in[6];
  const float* bpde = (const float*)d_in[7];
  float* out = (float*)d_out;

  char* ws = (char*)d_ws;
  size_t o = 0;
  auto alloc = [&](size_t bytes) {
    void* p = ws + o; o += (bytes + 255) & ~(size_t)255; return p;
  };
  const size_t MN = (size_t)4096 * 512;              // B*N x C
  bf16* seB   = (bf16*)alloc(MN * 2);                // later reused as att_se
  bf16* deB   = (bf16*)alloc(MN * 2);                // later reused as att_de
  bf16* WqkvT = (bf16*)alloc((size_t)1536 * 512 * 2);
  bf16* WvT   = (bf16*)alloc((size_t)512 * 512 * 2);
  bf16* WpseT = (bf16*)alloc((size_t)512 * 512 * 2);
  bf16* WpdeT = (bf16*)alloc((size_t)512 * 512 * 2);
  bf16* qkvB  = (bf16*)alloc((size_t)4096 * 1536 * 2);
  bf16* vdeB  = (bf16*)alloc(MN * 2);
  bf16* qB    = (bf16*)alloc(MN * 2);
  bf16* kB    = (bf16*)alloc(MN * 2);
  bf16* vsT   = (bf16*)alloc(MN * 2);
  bf16* vdT   = (bf16*)alloc(MN * 2);                // ~43 MiB total workspace

  dim3 blk(256);
  int nel = (int)MN;
  k_cvt<<<(nel + 255) / 256, blk, 0, stream>>>(se, seB, nel);
  k_cvt<<<(nel + 255) / 256, blk, 0, stream>>>(de, deB, nel);
  k_cvt_t<<<(512 * 1536 + 255) / 256, blk, 0, stream>>>(Wqkv, WqkvT, 512, 1536);
  k_cvt_t<<<(512 * 512 + 255) / 256, blk, 0, stream>>>(Wv,   WvT,   512, 512);
  k_cvt_t<<<(512 * 512 + 255) / 256, blk, 0, stream>>>(Wpse, WpseT, 512, 512);
  k_cvt_t<<<(512 * 512 + 255) / 256, blk, 0, stream>>>(Wpde, WpdeT, 512, 512);

  k_gemm<<<dim3(12, 32), blk, 0, stream>>>(seB, WqkvT, 4096, 1536, 512,
                                           nullptr, nullptr, qkvB);
  k_gemm<<<dim3(4, 32), blk, 0, stream>>>(deB, WvT, 4096, 512, 512,
                                          nullptr, nullptr, vdeB);

  k_repack<<<(B_ * H_ * N_ * D_ + 255) / 256, blk, 0, stream>>>(
      qkvB, vdeB, qB, kB, vsT, vdT);

  bf16* attse = seB;   // se/de bf16 copies are dead now; reuse their space
  bf16* attde = deB;
  k_attn<<<dim3(16, 32), blk, 0, stream>>>(qB, kB, vsT, vdT, attse, attde);

  k_gemm<<<dim3(4, 32), blk, 0, stream>>>(attse, WpseT, 4096, 512, 512,
                                          bpse, out, nullptr);
  k_gemm<<<dim3(4, 32), blk, 0, stream>>>(attde, WpdeT, 4096, 512, 512,
                                          bpde, out + MN, nullptr);
}
